// BiLinearAttn_87995289960997
// MI455X (gfx1250) — matrix operand order repaired and checked by reading, not compiled, not run on hardware
//
#include <hip/hip_runtime.h>
#include <math.h>

// Problem constants (match reference)
#define B_   16
#define LQ_  2048
#define LK_  2048
#define D1_  1024
#define D2_  1024

typedef __bf16 bf16;
typedef __attribute__((ext_vector_type(16))) __bf16 v16bf;
typedef __attribute__((ext_vector_type(8)))  __bf16 v8bf;
typedef __attribute__((ext_vector_type(4)))  __bf16 v4bf;
typedef __attribute__((ext_vector_type(8)))  float  v8f;
typedef __attribute__((ext_vector_type(4)))  float  v4f;
typedef __attribute__((ext_vector_type(4)))  int    v4i;
// Address-space-qualified int4 for the async global->LDS builtin signature.
typedef __attribute__((address_space(1))) v4i v4i_g;
typedef __attribute__((address_space(3))) v4i v4i_l;

// gfx1250 async global->LDS staging (ASYNCcnt path), with sync fallback.
#if defined(__has_builtin)
#if __has_builtin(__builtin_amdgcn_global_load_async_to_lds_b128) && \
    __has_builtin(__builtin_amdgcn_s_wait_asynccnt)
#define HAVE_ASYNC_LDS 1
#endif
#endif
#ifndef HAVE_ASYNC_LDS
#define HAVE_ASYNC_LDS 0
#endif

__device__ __forceinline__ v8f wmma_bf16(v16bf a, v16bf b, v8f c) {
  // D = A(16x32 bf16) x B(32x16 bf16) + C(16x16 f32)
  return __builtin_amdgcn_wmma_f32_16x16x32_bf16(
      /*neg_a=*/false, a, /*neg_b=*/false, b,
      /*c_mod=*/(short)0, c, /*reuse_a=*/false, /*reuse_b=*/false);
}

// ---- LDS staging of shared 128x32 bf16 B-tiles -----------------------------
#define TPAD 40                    // padded row stride (80 B = 20 banks)
#define BTILE_ELEMS (128 * TPAD)   // one buffer

__device__ __forceinline__ void async_copy16(const bf16* gsrc, bf16* ldst) {
#if HAVE_ASYNC_LDS
  __builtin_amdgcn_global_load_async_to_lds_b128(
      (v4i_g*)gsrc, (v4i_l*)ldst, 0, 0);
#else
  *(v8bf*)ldst = *(const v8bf*)gsrc;
#endif
}

__device__ __forceinline__ void wait_async_lds() {
#if HAVE_ASYNC_LDS
  __builtin_amdgcn_s_wait_asynccnt(0);
#endif
}

// Stage 128 rows x 32 cols bf16 from row-major [.,ld] global into tile[128][TPAD].
// 256 threads: row = tid>>1, col half = (tid&1)*16; 2x16B async copies per lane.
__device__ __forceinline__ void stage_b_tile(bf16* tile, const bf16* __restrict__ gbase,
                                             int ld, int n0, int k0, int tid) {
  const int row  = tid >> 1;
  const int half = (tid & 1) << 4;
  const bf16* src = gbase + (size_t)(n0 + row) * ld + k0 + half;
  bf16* dst = tile + row * TPAD + half;
  async_copy16(src, dst);
  async_copy16(src + 8, dst + 8);
}

// ---- Fragment loaders (wave32 layouts per CDNA5 ISA 7.12.2) ----------------

// A 16x32 bf16 fragment from row-major [rows][ld] bf16.
__device__ __forceinline__ v16bf load_a_bf16(const bf16* __restrict__ base, int ld,
                                             int row0, int k0, int lane) {
  const int r  = row0 + (lane & 15);
  const int kh = k0 + ((lane >> 4) << 3);
  const bf16* p = base + (size_t)r * ld + kh;
  v8bf lo = *(const v8bf*)(p);
  v8bf hi = *(const v8bf*)(p + 16);
  v16bf a;
#pragma unroll
  for (int i = 0; i < 8; ++i) { a[i] = lo[i]; a[i + 8] = hi[i]; }
  return a;
}

// Same A fragment but converting from fp32 source on the fly.
__device__ __forceinline__ v16bf load_a_f32(const float* __restrict__ base, int ld,
                                            int row0, int k0, int lane) {
  const int r  = row0 + (lane & 15);
  const int kh = k0 + ((lane >> 4) << 3);
  const float* p = base + (size_t)r * ld + kh;
  v4f x0 = *(const v4f*)(p);
  v4f x1 = *(const v4f*)(p + 4);
  v4f x2 = *(const v4f*)(p + 16);
  v4f x3 = *(const v4f*)(p + 20);
  v16bf a;
#pragma unroll
  for (int i = 0; i < 4; ++i) {
    a[i]      = (bf16)x0[i];
    a[i + 4]  = (bf16)x1[i];
    a[i + 8]  = (bf16)x2[i];
    a[i + 12] = (bf16)x3[i];
  }
  return a;
}

// B 32x16 fragment from a staged LDS tile (rows = N, 32 K-cols, TPAD stride).
// lane&15 = N col; half-wave h selects K 8h..8h+7 and 16+8h..16+8h+7.
__device__ __forceinline__ v16bf load_b_lds(const bf16* tile, int n0, int lane) {
  const bf16* p = tile + (n0 + (lane & 15)) * TPAD + ((lane >> 4) << 3);
  v8bf lo = *(const v8bf*)(p);
  v8bf hi = *(const v8bf*)(p + 16);
  v16bf b;
#pragma unroll
  for (int i = 0; i < 8; ++i) { b[i] = lo[i]; b[i + 8] = hi[i]; }
  return b;
}

// ---- Prep kernels ----------------------------------------------------------

// Generic fp32 -> bf16 cast (used for keys and W); raw cast keeps -inf.
__global__ void cast_f32_bf16_kernel(const float4* __restrict__ src,
                                     v4bf* __restrict__ dst, int n4) {
  int i = blockIdx.x * blockDim.x + threadIdx.x;
  if (i >= n4) return;
  float4 v = src[i];
  v4bf o;
  o[0] = (bf16)v.x; o[1] = (bf16)v.y; o[2] = (bf16)v.z; o[3] = (bf16)v.w;
  dst[i] = o;
}

// Vt[b][d][k] = (keys[b][k][d] == -inf) ? 0 : keys[b][k][d], bf16, transposed.
__global__ void build_vt_kernel(const float* __restrict__ keys,
                                bf16* __restrict__ vt) {
  __shared__ bf16 tile[32][33];
  const int b = blockIdx.z, k0 = blockIdx.x * 32, d0 = blockIdx.y * 32;
  const float* src = keys + ((size_t)b * LK_ + k0) * D2_ + d0;
  for (int i = threadIdx.y; i < 32; i += 8) {
    float v = src[(size_t)i * D2_ + threadIdx.x];
    tile[i][threadIdx.x] = (v == -__builtin_inff()) ? (bf16)0.0f : (bf16)v;
  }
  __syncthreads();
  bf16* dst = vt + ((size_t)b * D2_ + d0) * LK_ + k0;
  for (int i = threadIdx.y; i < 32; i += 8) {
    dst[(size_t)i * LK_ + threadIdx.x] = tile[threadIdx.x][i];
  }
}

// ---- Q projection: Qp = Q * W^T + bias (bf16 out, fp32 accumulate) ---------
// grid (LQ/64, D2/128, B), block 256 (4 waves M x 2 waves N), wave tile 16x64.
// W (bf16) staged per k-step through LDS (async, double buffered).
__global__ __launch_bounds__(256)
void qproj_kernel(const float* __restrict__ Q, const bf16* __restrict__ Wbf,
                  const float* __restrict__ bias, bf16* __restrict__ Qp) {
  __shared__ bf16 btile[2][BTILE_ELEMS];
  const int tid = threadIdx.x, lane = tid & 31, wid = tid >> 5;
  const int wm = wid & 3, wn = wid >> 2;
  const int row0 = blockIdx.x * 64 + wm * 16;
  const int ncol = blockIdx.y * 128;
  const float* Qb = Q + (size_t)blockIdx.z * LQ_ * D1_;

  v8f acc[4] = {};
  stage_b_tile(btile[0], Wbf, D1_, ncol, 0, tid);
  for (int k = 0; k < D1_; k += 32) {
    const int cur = (k >> 5) & 1;
    wait_async_lds();
    __syncthreads();
    if (k + 32 < D1_)
      stage_b_tile(btile[cur ^ 1], Wbf, D1_, ncol, k + 32, tid);
    v16bf a = load_a_f32(Qb, D1_, row0, k, lane);
#pragma unroll
    for (int t = 0; t < 4; ++t)
      acc[t] = wmma_bf16(a, load_b_lds(btile[cur], wn * 64 + t * 16, lane), acc[t]);
  }

  bf16* outb = Qp + (size_t)blockIdx.z * LQ_ * D2_;
  const int rbase = row0 + ((lane >> 4) << 3);
  const int cl = lane & 15;
#pragma unroll
  for (int t = 0; t < 4; ++t) {
    const int c = ncol + wn * 64 + t * 16 + cl;
    const float bv = bias[c];
#pragma unroll
    for (int r = 0; r < 8; ++r)
      outb[(size_t)(rbase + r) * D2_ + c] = (bf16)(acc[t][r] + bv);
  }
}

// ---- Pass 1: softmax statistics (row max m, row sumexp l) ------------------
// grid (LQ/128, B), block 256; wave = one 16-row q-tile; all 8 waves share the
// async-staged K tile (128 keys x 32 d per step).
__global__ __launch_bounds__(256)
void attn_stats_kernel(const bf16* __restrict__ Qp, const bf16* __restrict__ Kbf,
                       float* __restrict__ mbuf, float* __restrict__ lbuf) {
  __shared__ bf16 ktile[2][BTILE_ELEMS];
  const int tid = threadIdx.x, lane = tid & 31, wid = tid >> 5;
  const int b = blockIdx.y;
  const int row0 = blockIdx.x * 128 + wid * 16;
  const bf16* Qb = Qp + (size_t)b * LQ_ * D2_;
  const bf16* Kb = Kbf + (size_t)b * LK_ * D2_;
  const float NEG_INF = -__builtin_inff();

  float m[8], l[8];
#pragma unroll
  for (int r = 0; r < 8; ++r) { m[r] = NEG_INF; l[r] = 0.0f; }

  for (int kt = 0; kt < LK_; kt += 128) {
    v8f s[8] = {};
    stage_b_tile(ktile[0], Kb, D2_, kt, 0, tid);
    for (int k = 0; k < D2_; k += 32) {
      const int cur = (k >> 5) & 1;
      wait_async_lds();
      __syncthreads();
      if (k + 32 < D2_)
        stage_b_tile(ktile[cur ^ 1], Kb, D2_, kt, k + 32, tid);
      v16bf a = load_a_bf16(Qb, D2_, row0, k, lane);
#pragma unroll
      for (int t = 0; t < 8; ++t)
        s[t] = wmma_bf16(a, load_b_lds(ktile[cur], t * 16, lane), s[t]);
    }
#pragma unroll
    for (int r = 0; r < 8; ++r) {
      float vals[8];
      float rowmax = NEG_INF;
#pragma unroll
      for (int t = 0; t < 8; ++t) {
        float v = s[t][r];
        v = (v != v) ? NEG_INF : v;   // NaN -> -inf (reference semantics)
        vals[t] = v;
        rowmax = fmaxf(rowmax, v);
      }
#pragma unroll
      for (int off = 1; off < 16; off <<= 1)
        rowmax = fmaxf(rowmax, __shfl_xor(rowmax, off, 32));
      const float newm = fmaxf(m[r], rowmax);
      float esum = 0.0f;
#pragma unroll
      for (int t = 0; t < 8; ++t) esum += __expf(vals[t] - newm);
#pragma unroll
      for (int off = 1; off < 16; off <<= 1)
        esum += __shfl_xor(esum, off, 32);
      l[r] = l[r] * __expf(m[r] - newm) + esum;
      m[r] = newm;
    }
  }

  if ((lane & 15) == 0) {
    const int rb = row0 + ((lane >> 4) << 3);
#pragma unroll
    for (int r = 0; r < 8; ++r) {
      mbuf[(size_t)b * LQ_ + rb + r] = m[r];
      lbuf[(size_t)b * LQ_ + rb + r] = l[r];
    }
  }
}

// ---- Pass 2: recompute S, emit P = exp(S - m) / l as bf16 ------------------
__global__ __launch_bounds__(256)
void attn_probs_kernel(const bf16* __restrict__ Qp, const bf16* __restrict__ Kbf,
                       const float* __restrict__ mbuf, const float* __restrict__ lbuf,
                       bf16* __restrict__ P) {
  __shared__ bf16 ktile[2][BTILE_ELEMS];
  const int tid = threadIdx.x, lane = tid & 31, wid = tid >> 5;
  const int b = blockIdx.y;
  const int row0 = blockIdx.x * 128 + wid * 16;
  const bf16* Qb = Qp + (size_t)b * LQ_ * D2_;
  const bf16* Kb = Kbf + (size_t)b * LK_ * D2_;
  bf16* Pb = P + (size_t)b * LQ_ * LK_;
  const float NEG_INF = -__builtin_inff();

  const int rbase = row0 + ((lane >> 4) << 3);
  const int cl = lane & 15;
  float mr[8], lr[8];
#pragma unroll
  for (int r = 0; r < 8; ++r) {
    mr[r] = mbuf[(size_t)b * LQ_ + rbase + r];
    lr[r] = 1.0f / lbuf[(size_t)b * LQ_ + rbase + r];
  }

  for (int kt = 0; kt < LK_; kt += 128) {
    v8f s[8] = {};
    stage_b_tile(ktile[0], Kb, D2_, kt, 0, tid);
    for (int k = 0; k < D2_; k += 32) {
      const int cur = (k >> 5) & 1;
      wait_async_lds();
      __syncthreads();
      if (k + 32 < D2_)
        stage_b_tile(ktile[cur ^ 1], Kb, D2_, kt, k + 32, tid);
      v16bf a = load_a_bf16(Qb, D2_, row0, k, lane);
#pragma unroll
      for (int t = 0; t < 8; ++t)
        s[t] = wmma_bf16(a, load_b_lds(ktile[cur], t * 16, lane), s[t]);
    }
#pragma unroll
    for (int t = 0; t < 8; ++t) {
      const int c = kt + t * 16 + cl;
#pragma unroll
      for (int r = 0; r < 8; ++r) {
        float v = s[t][r];
        v = (v != v) ? NEG_INF : v;
        const float p = __expf(v - mr[r]) * lr[r];
        Pb[(size_t)(rbase + r) * LK_ + c] = (bf16)p;
      }
    }
  }
}

// ---- Output GEMM: O = P (LQxLK) * V (LKxD2), V read via Vt[D2][LK] ---------
// grid (LQ/64, D2/128, B), block 256, wave tile 16x64; Vt tile async-staged.
__global__ __launch_bounds__(256)
void attn_out_kernel(const bf16* __restrict__ P, const bf16* __restrict__ Vt,
                     float* __restrict__ O) {
  __shared__ bf16 btile[2][BTILE_ELEMS];
  const int tid = threadIdx.x, lane = tid & 31, wid = tid >> 5;
  const int wm = wid & 3, wn = wid >> 2;
  const int row0 = blockIdx.x * 64 + wm * 16;
  const int ncol = blockIdx.y * 128;
  const bf16* Pb  = P + (size_t)blockIdx.z * LQ_ * LK_;
  const bf16* Vtb = Vt + (size_t)blockIdx.z * D2_ * LK_;

  v8f acc[4] = {};
  stage_b_tile(btile[0], Vtb, LK_, ncol, 0, tid);
  for (int k = 0; k < LK_; k += 32) {
    const int cur = (k >> 5) & 1;
    wait_async_lds();
    __syncthreads();
    if (k + 32 < LK_)
      stage_b_tile(btile[cur ^ 1], Vtb, LK_, ncol, k + 32, tid);
    v16bf a = load_a_bf16(Pb, LK_, row0, k, lane);
#pragma unroll
    for (int t = 0; t < 4; ++t)
      acc[t] = wmma_bf16(a, load_b_lds(btile[cur], wn * 64 + t * 16, lane), acc[t]);
  }

  float* Ob = O + (size_t)blockIdx.z * LQ_ * D2_;
  const int rbase = row0 + ((lane >> 4) << 3);
  const int cl = lane & 15;
#pragma unroll
  for (int t = 0; t < 4; ++t) {
    const int c = ncol + wn * 64 + t * 16 + cl;
#pragma unroll
    for (int r = 0; r < 8; ++r)
      Ob[(size_t)(rbase + r) * D2_ + c] = acc[t][r];
  }
}

// ---- Host-side launcher ----------------------------------------------------
extern "C" void kernel_launch(void* const* d_in, const int* in_sizes, int n_in,
                              void* d_out, int out_size, void* d_ws, size_t ws_size,
                              hipStream_t stream) {
  const float* keys    = (const float*)d_in[0];
  const float* queries = (const float*)d_in[1];
  const float* W       = (const float*)d_in[2];
  const float* bias    = (const float*)d_in[3];
  float* out = (float*)d_out;
  (void)in_sizes; (void)n_in; (void)out_size; (void)ws_size;

  char* ws = (char*)d_ws;
  size_t off = 0;
  bf16* Qp  = (bf16*)(ws + off); off += (size_t)B_ * LQ_ * D2_ * sizeof(bf16);  //  64 MB
  bf16* Kbf = (bf16*)(ws + off); off += (size_t)B_ * LK_ * D2_ * sizeof(bf16);  //  64 MB
  bf16* Vt  = (bf16*)(ws + off); off += (size_t)B_ * D2_ * LK_ * sizeof(bf16);  //  64 MB
  bf16* P   = (bf16*)(ws + off); off += (size_t)B_ * LQ_ * LK_ * sizeof(bf16);  // 128 MB
  bf16* Wbf = (bf16*)(ws + off); off += (size_t)D2_ * D1_ * sizeof(bf16);       //   2 MB
  float* mbuf = (float*)(ws + off); off += (size_t)B_ * LQ_ * sizeof(float);
  float* lbuf = (float*)(ws + off); off += (size_t)B_ * LQ_ * sizeof(float);

  {
    const int n4 = (B_ * LK_ * D2_) / 4;
    cast_f32_bf16_kernel<<<dim3(n4 / 256), dim3(256), 0, stream>>>(
        (const float4*)keys, (v4bf*)Kbf, n4);
  }
  {
    const int n4 = (D2_ * D1_) / 4;
    cast_f32_bf16_kernel<<<dim3(n4 / 256), dim3(256), 0, stream>>>(
        (const float4*)W, (v4bf*)Wbf, n4);
  }
  build_vt_kernel<<<dim3(LK_ / 32, D2_ / 32, B_), dim3(32, 8), 0, stream>>>(keys, Vt);
  qproj_kernel<<<dim3(LQ_ / 64, D2_ / 128, B_), dim3(256), 0, stream>>>(queries, Wbf, bias, Qp);
  attn_stats_kernel<<<dim3(LQ_ / 128, B_), dim3(256), 0, stream>>>(Qp, Kbf, mbuf, lbuf);
  attn_probs_kernel<<<dim3(LQ_ / 128, B_), dim3(256), 0, stream>>>(Qp, Kbf, mbuf, lbuf, P);
  attn_out_kernel<<<dim3(LQ_ / 64, D2_ / 128, B_), dim3(256), 0, stream>>>(P, Vt, out);
}
